// VectorQuantizer_5720896438809
// MI455X (gfx1250) — matrix operand order repaired and checked by reading, not compiled, not run on hardware
//
#include <hip/hip_runtime.h>
#include <math.h>

#define N_ROWS   32768
#define DIMS     256
#define KCODES   4096
#define NELEM    (N_ROWS * DIMS)   // 8388608
#define NTILES   (KCODES / 16)     // 256 column tiles
#define BPAD     264               // padded LDS row stride (elements)

typedef __attribute__((ext_vector_type(16))) __bf16 v16bf;
typedef __attribute__((ext_vector_type(8)))  __bf16 v8bf;
typedef __attribute__((ext_vector_type(8)))  float  v8f;

#if defined(__has_builtin)
#  if __has_builtin(__builtin_amdgcn_sched_barrier)
#    define SCHED_FENCE() __builtin_amdgcn_sched_barrier(0)
#  endif
#endif
#ifndef SCHED_FENCE
#  define SCHED_FENCE()
#endif

// ---------------------------------------------------------------------------
// Kernel 1: codebook -> bf16, halfnorm[k] = 0.5*|e_k|^2, zero histogram.
// ---------------------------------------------------------------------------
__global__ void __launch_bounds__(256) vq_prep(const float* __restrict__ codebook,
                                               __bf16* __restrict__ cb,
                                               float* __restrict__ halfn,
                                               int* __restrict__ counts) {
    const int k = blockIdx.x;
    const int t = threadIdx.x;
    float c = codebook[k * DIMS + t];
    cb[k * DIMS + t] = (__bf16)c;
    float v = c * c;
    __shared__ float sm[8];
    const int lane = t & 31, w = t >> 5;
    #pragma unroll
    for (int o = 16; o > 0; o >>= 1) v += __shfl_down(v, o, 32);
    if (lane == 0) sm[w] = v;
    __syncthreads();
    if (t == 0) {
        float s = 0.f;
        #pragma unroll
        for (int i = 0; i < 8; ++i) s += sm[i];
        halfn[k] = 0.5f * s;
        counts[k] = 0;   // re-zero histogram every launch
    }
}

// ---------------------------------------------------------------------------
// Kernel 2: fused GEMM + argmax, LDS double buffer + 2-deep global pipeline.
// argmin_k (|x|^2 + |e_k|^2 - 2 x.e_k)  ==  argmax_k (x.e_k - 0.5|e_k|^2)
// Entering iter kt: LDS buf[kt&1] = tile kt, regs gcur = tile kt+1, hcur set.
// ---------------------------------------------------------------------------
__global__ void __launch_bounds__(256) vq_gemm_argmax(const float* __restrict__ x,
                                                      const __bf16* __restrict__ cb,
                                                      const float* __restrict__ halfn,
                                                      int* __restrict__ idxOut) {
    __shared__ __bf16 Bs[2][16 * BPAD];   // 2 x 8.25 KB double buffer

    const int t       = threadIdx.x;
    const int lane    = t & 31;
    const int wave    = t >> 5;
    const int rowBase = blockIdx.x * 128 + wave * 16;
    const int hi      = (lane & 16) ? 1 : 0;         // half-wave
    const int m       = rowBase + (lane & 15);

    // Staging assignment: thread t copies one 32-byte segment of each tile.
    const int sCol = t >> 4;                         // column within tile
    const int sSeg = t & 15;                         // 16-elem segment
    const __bf16* sgp = cb + (size_t)sCol * DIMS + sSeg * 16;
    __bf16* slp0 = &Bs[0][sCol * BPAD + sSeg * 16];
    __bf16* slp1 = &Bs[1][sCol * BPAD + sSeg * 16];

    // --- Load + convert this lane's A tile (ISA 16-bit A layout):
    // lanes 0-15 : K = 0..7 , 16..23   lanes 16-31 : K = 8..15, 24..31
    v16bf a[8];
    {
        const float* xr = x + (size_t)m * DIMS + (hi ? 8 : 0);
        #pragma unroll
        for (int c = 0; c < 8; ++c) {
            const float* p = xr + c * 32;
            #pragma unroll
            for (int i = 0; i < 8; ++i) a[c][i]     = (__bf16)p[i];
            #pragma unroll
            for (int i = 0; i < 8; ++i) a[c][8 + i] = (__bf16)p[16 + i];
        }
    }

    // --- Prologue: tile 0 -> LDS buf0; tile 1 -> gcur; hcur = halfn(tile 0).
    v8bf gc0, gc1;
    float hcur;
    {
        v8bf t0a = *(const v8bf*)sgp;
        v8bf t0b = *(const v8bf*)(sgp + 8);
        const __bf16* g1p = sgp + (size_t)16 * DIMS;
        gc0 = *(const v8bf*)g1p;
        gc1 = *(const v8bf*)(g1p + 8);
        hcur = halfn[lane & 15];
        *(v8bf*)slp0       = t0a;
        *(v8bf*)(slp0 + 8) = t0b;
    }
    __syncthreads();

    float bestS[8];
    int   bestI[8];
    #pragma unroll
    for (int v = 0; v < 8; ++v) { bestS[v] = -3.4e38f; bestI[v] = 0; }

    const __bf16* lrd = &Bs[0][(lane & 15) * BPAD + hi * 8];
    const int bufStride = 16 * BPAD;

    for (int kt = 0; kt < NTILES; ++kt) {
        // ---- Issue prefetches whose consumers are one iteration away.
        v8bf gn0, gn1;
        float hnext = 0.f;
        if (kt + 2 < NTILES) {
            const __bf16* gp = sgp + (size_t)(kt + 2) * 16 * DIMS;
            gn0 = *(const v8bf*)gp;
            gn1 = *(const v8bf*)(gp + 8);
        }
        if (kt + 1 < NTILES) {
            hnext = halfn[(kt + 1) * 16 + (lane & 15)];
            // gcur (loaded last iteration) -> LDS buffer for tile kt+1.
            __bf16* sl = ((kt + 1) & 1) ? slp1 : slp0;
            *(v8bf*)sl       = gc0;
            *(v8bf*)(sl + 8) = gc1;
        }
        SCHED_FENCE();   // keep prefetch + stage above the compute group

        // ---- Batch-load all 8 B chunks from LDS, then the WMMA chain.
        const __bf16* lb = lrd + (kt & 1) * bufStride;
        v16bf b[8];
        #pragma unroll
        for (int c = 0; c < 8; ++c) {
            v8bf blo = *(const v8bf*)(lb + c * 32);
            v8bf bhi = *(const v8bf*)(lb + c * 32 + 16);
            b[c] = __builtin_shufflevector(blo, bhi,
                                           0,1,2,3,4,5,6,7,8,9,10,11,12,13,14,15);
        }
        v8f acc0 = {};
        v8f acc1 = {};
        #pragma unroll
        for (int c = 0; c < 8; c += 2) {
            acc0 = __builtin_amdgcn_wmma_f32_16x16x32_bf16(
                       false, a[c],     false, b[c],     (short)0, acc0, false, false);
            acc1 = __builtin_amdgcn_wmma_f32_16x16x32_bf16(
                       false, a[c + 1], false, b[c + 1], (short)0, acc1, false, false);
        }

        const int col = kt * 16 + (lane & 15);
        #pragma unroll
        for (int v = 0; v < 8; ++v) {
            float s = (acc0[v] + acc1[v]) - hcur;
            if (s > bestS[v]) { bestS[v] = s; bestI[v] = col; }
        }

        __syncthreads();          // closes RAW (stores->readers) + WAR windows
        gc0 = gn0; gc1 = gn1;     // rotate pipeline registers
        hcur = hnext;
    }

    // C/D layout: component v => row v (lanes 0-15) / row v+8 (lanes 16-31);
    // columns live across the 16 lanes of each half -> reduce within width 16.
    #pragma unroll
    for (int o = 8; o > 0; o >>= 1) {
        #pragma unroll
        for (int v = 0; v < 8; ++v) {
            float os = __shfl_xor(bestS[v], o, 16);
            int   oi = __shfl_xor(bestI[v], o, 16);
            if (os > bestS[v] || (os == bestS[v] && oi < bestI[v])) {
                bestS[v] = os; bestI[v] = oi;
            }
        }
    }
    if ((lane & 15) == 0) {
        const int rb = rowBase + (hi ? 8 : 0);
        #pragma unroll
        for (int v = 0; v < 8; ++v) idxOut[rb + v] = bestI[v];
    }
}

// ---------------------------------------------------------------------------
// Kernel 3: gather quantized rows + per-block sum((q-x)^2) partials.
// ---------------------------------------------------------------------------
__global__ void __launch_bounds__(256) vq_gather_loss(const float* __restrict__ x,
                                                      const float* __restrict__ codebook,
                                                      const int* __restrict__ idxArr,
                                                      float* __restrict__ out,
                                                      float* __restrict__ partials) {
    const int t = threadIdx.x;
    float s = 0.f;
    #pragma unroll
    for (int i = 0; i < 8; ++i) {
        const int e   = blockIdx.x * 8192 + i * 1024 + t;
        const int row = e >> 8;
        const int d   = e & 255;
        const int k   = idxArr[row];
        const float q  = codebook[k * DIMS + d];
        const float xv = x[e];
        out[e] = q;
        const float df = q - xv;
        s += df * df;
    }
    __shared__ float sm[8];
    const int lane = t & 31, w = t >> 5;
    #pragma unroll
    for (int o = 16; o > 0; o >>= 1) s += __shfl_down(s, o, 32);
    if (lane == 0) sm[w] = s;
    __syncthreads();
    if (t == 0) {
        float r = 0.f;
        #pragma unroll
        for (int i = 0; i < 8; ++i) r += sm[i];
        partials[blockIdx.x] = r;
    }
}

// ---------------------------------------------------------------------------
// Kernel 4: assignment histogram (integer atomics -> order independent).
// ---------------------------------------------------------------------------
__global__ void __launch_bounds__(256) vq_hist(const int* __restrict__ idxArr,
                                               int* __restrict__ counts) {
    const int i = blockIdx.x * 256 + threadIdx.x;
    atomicAdd(&counts[idxArr[i]], 1);
}

// ---------------------------------------------------------------------------
// Kernel 5: finalize vq_loss and perplexity.
// ---------------------------------------------------------------------------
__global__ void __launch_bounds__(256) vq_finalize(const float* __restrict__ partials,
                                                   const int* __restrict__ counts,
                                                   float* __restrict__ outScalars) {
    const int t = threadIdx.x;
    float ls = 0.f, ent = 0.f;
    for (int i = t; i < 1024; i += 256) ls += partials[i];
    for (int k = t; k < KCODES; k += 256) {
        const float p = (float)counts[k] * (1.0f / (float)N_ROWS);
        ent += p * logf(p + 1e-10f);
    }
    __shared__ float sm1[8], sm2[8];
    const int lane = t & 31, w = t >> 5;
    #pragma unroll
    for (int o = 16; o > 0; o >>= 1) {
        ls  += __shfl_down(ls,  o, 32);
        ent += __shfl_down(ent, o, 32);
    }
    if (lane == 0) { sm1[w] = ls; sm2[w] = ent; }
    __syncthreads();
    if (t == 0) {
        float L = 0.f, E = 0.f;
        #pragma unroll
        for (int i = 0; i < 8; ++i) { L += sm1[i]; E += sm2[i]; }
        outScalars[0] = 1.25f * L / (float)NELEM;  // codebook + 0.25*commitment
        outScalars[1] = expf(-E);                  // perplexity
    }
}

// ---------------------------------------------------------------------------
extern "C" void kernel_launch(void* const* d_in, const int* in_sizes, int n_in,
                              void* d_out, int out_size, void* d_ws, size_t ws_size,
                              hipStream_t stream) {
    const float* x        = (const float*)d_in[0];   // (32,32,32,256) fp32
    const float* codebook = (const float*)d_in[1];   // (4096,256) fp32
    float* out = (float*)d_out;                      // quantized | vq_loss | perplexity

    char* ws = (char*)d_ws;
    size_t off = 0;
    __bf16* cb     = (__bf16*)(ws + off); off += (size_t)KCODES * DIMS * sizeof(__bf16); // 2 MB
    float*  halfn  = (float*) (ws + off); off += (size_t)KCODES * sizeof(float);         // 16 KB
    int*    idxArr = (int*)   (ws + off); off += (size_t)N_ROWS * sizeof(int);           // 128 KB
    int*    counts = (int*)   (ws + off); off += (size_t)KCODES * sizeof(int);           // 16 KB
    float*  parts  = (float*) (ws + off);                                                // 4 KB

    vq_prep       <<<KCODES,       256, 0, stream>>>(codebook, cb, halfn, counts);
    vq_gemm_argmax<<<N_ROWS / 128, 256, 0, stream>>>(x, cb, halfn, idxArr);
    vq_gather_loss<<<NELEM / 8192, 256, 0, stream>>>(x, codebook, idxArr, out, parts);
    vq_hist       <<<N_ROWS / 256, 256, 0, stream>>>(idxArr, counts);
    vq_finalize   <<<1,            256, 0, stream>>>(parts, counts, out + NELEM);
}
